// MHA_89455578841228
// MI455X (gfx1250) — matrix operand order repaired and checked by reading, not compiled, not run on hardware
//
#include <hip/hip_runtime.h>
#include <hip/hip_bf16.h>

// B=2, T=2048, D=2048, H=16, HD=128.
// Pipeline: [split x + weights into bf16 hi/lo] -> [bf16x3 async-staged GEMM qkv]
//        -> [RoPE in-place] -> [async-staged f32-WMMA flash attention -> bf16 hi/lo]
//        -> [bf16x3 async-staged GEMM out].

typedef __attribute__((ext_vector_type(2)))  float  v2f;
typedef __attribute__((ext_vector_type(8)))  float  v8f;
typedef __attribute__((ext_vector_type(8)))  __bf16 v8bf;
typedef __attribute__((ext_vector_type(16))) __bf16 v16bf;
typedef __attribute__((ext_vector_type(4)))  int    v4i;

__device__ inline v8f wmma_f32(v2f a, v2f b, v8f c) {
    return __builtin_amdgcn_wmma_f32_16x16x4_f32(
        false, a, false, b, (short)0, c, false, false);
}
__device__ inline v8f wmma_bf16(v16bf a, v16bf b, v8f c) {
    return __builtin_amdgcn_wmma_f32_16x16x32_bf16(
        false, a, false, b, (short)0, c, false, false);
}

// ---- async global->LDS 16B copy (CDNA5), with synchronous fallback ---------
#if __has_builtin(__builtin_amdgcn_global_load_async_to_lds_b128) && \
    __has_builtin(__builtin_amdgcn_s_wait_asynccnt)
#define ASYNC_LDS 1
#endif

__device__ inline void cp16_g2l(const void* g, void* l) {
#ifdef ASYNC_LDS
    __builtin_amdgcn_global_load_async_to_lds_b128(
        (__attribute__((address_space(1))) v4i*)g,
        (__attribute__((address_space(3))) v4i*)l, 0, 0);
#else
    *(v4i*)l = *(const v4i*)g;
#endif
}
__device__ inline void wait_cp() {
#ifdef ASYNC_LDS
    __builtin_amdgcn_s_wait_asynccnt(0);
#endif
}

// ---------------------------------------------------------------------------
// Split natural-layout f32 matrix into bf16 hi/lo (same layout).
// ---------------------------------------------------------------------------
__global__ __launch_bounds__(256) void split_x(
    const float* __restrict__ X, __bf16* __restrict__ Xh, __bf16* __restrict__ Xl)
{
    const size_t idx = (size_t)blockIdx.x * blockDim.x + threadIdx.x;
    float w = X[idx];
    __bf16 h = (__bf16)w;
    Xh[idx] = h;
    Xl[idx] = (__bf16)(w - (float)h);
}

// ---------------------------------------------------------------------------
// Split W[K,N] f32 into transposed bf16 hi/lo: Wt{h,l}[N,K].
// ---------------------------------------------------------------------------
__global__ __launch_bounds__(256) void split_w_t(
    const float* __restrict__ W, __bf16* __restrict__ Wth,
    __bf16* __restrict__ Wtl, int K, int N)
{
    const size_t idx = (size_t)blockIdx.x * blockDim.x + threadIdx.x; // n*K + k
    const int k = (int)(idx % K);
    const int n = (int)(idx / K);
    float w = W[(size_t)k * N + n];
    __bf16 h = (__bf16)w;
    Wth[idx] = h;
    Wtl[idx] = (__bf16)(w - (float)h);
}

// ---------------------------------------------------------------------------
// bf16x3 GEMM: C[M,N] = (Ah+Al)[M,K] * (Bh+Bl)^T[N,K], dropping Al*Bl.
// All operands pre-split bf16. Tiles staged async into double-buffered LDS.
// Block = 256 threads (8 waves); tile 128x128; BK=32; wave = 32x64 (2x4).
// ---------------------------------------------------------------------------
__global__ __launch_bounds__(256) void gemm_bf16x3(
    const __bf16* __restrict__ Ahg, const __bf16* __restrict__ Alg,
    const __bf16* __restrict__ Bhg, const __bf16* __restrict__ Blg,
    float* __restrict__ C, int M, int N, int K)
{
    // [buf][operand: Ah,Al,Bh,Bl][128 rows * 32 k] bf16 = 64 KB total.
    __shared__ __align__(32) __bf16 Ls[2][4][128 * 32];

    const int tid  = threadIdx.x;
    const int lane = tid & 31;
    const int wave = tid >> 5;
    const int wm   = wave & 3;
    const int wn   = wave >> 2;
    const int half = lane >> 4;
    const int l15  = lane & 15;
    const int bm   = blockIdx.y * 128;
    const int bn   = blockIdx.x * 128;
    const int nT   = K >> 5;

    v8f acc[2][4];
#pragma unroll
    for (int i = 0; i < 2; ++i)
#pragma unroll
        for (int j = 0; j < 4; ++j) acc[i][j] = (v8f){};

    // Stage one 4-operand K-slab (each 128x32 bf16, rows of 64B = 4x16B).
    auto stage = [&](int kt, int buf) {
        const int k0 = kt * 32;
#pragma unroll
        for (int i = 0; i < 2; ++i) {
            int e = tid + i * 256;
            int r = e >> 2, c = (e & 3) * 8;
            size_t ga = (size_t)(bm + r) * K + k0 + c;
            size_t gb = (size_t)(bn + r) * K + k0 + c;
            int lo = r * 32 + c;
            cp16_g2l(Ahg + ga, &Ls[buf][0][lo]);
            cp16_g2l(Alg + ga, &Ls[buf][1][lo]);
            cp16_g2l(Bhg + gb, &Ls[buf][2][lo]);
            cp16_g2l(Blg + gb, &Ls[buf][3][lo]);
        }
    };

    stage(0, 0);
    for (int t = 0; t < nT; ++t) {
        const int buf = t & 1;
        wait_cp();          // this wave's async copies for tile t complete
        __syncthreads();    // all waves' copies done; prior compute done
        if (t + 1 < nT) stage(t + 1, buf ^ 1);  // overlap DMA with compute

        const __bf16* Ah = Ls[buf][0];
        const __bf16* Al = Ls[buf][1];
        const __bf16* Bh = Ls[buf][2];
        const __bf16* Bl = Ls[buf][3];

        // A fragment (16-bit A 16x32): lanes 0-15 hold K0-15;
        // lanes 16-31 hold K16-31 -> two 16B LDS loads combined.
        v16bf afh[2], afl[2], bfh[4], bfl[4];
#pragma unroll
        for (int i = 0; i < 2; ++i) {
            const __bf16* pa = &Ah[(wm * 32 + i * 16 + l15) * 32 + half * 16];
            v8bf p0 = *(const v8bf*)pa;
            v8bf p1 = *(const v8bf*)(pa + 8);
            afh[i] = __builtin_shufflevector(p0, p1, 0,1,2,3,4,5,6,7,8,9,10,11,12,13,14,15);
            const __bf16* pl = &Al[(wm * 32 + i * 16 + l15) * 32 + half * 16];
            v8bf q0 = *(const v8bf*)pl;
            v8bf q1 = *(const v8bf*)(pl + 8);
            afl[i] = __builtin_shufflevector(q0, q1, 0,1,2,3,4,5,6,7,8,9,10,11,12,13,14,15);
        }
        // B fragment (16-bit B 32x16): lanes 0-15 = K0-15, lanes 16-31 = K16-31.
#pragma unroll
        for (int j = 0; j < 4; ++j) {
            bfh[j] = *(const v16bf*)&Bh[(wn * 64 + j * 16 + l15) * 32 + half * 16];
            bfl[j] = *(const v16bf*)&Bl[(wn * 64 + j * 16 + l15) * 32 + half * 16];
        }
#pragma unroll
        for (int i = 0; i < 2; ++i)
#pragma unroll
            for (int j = 0; j < 4; ++j) {
                acc[i][j] = wmma_bf16(afh[i], bfh[j], acc[i][j]);
                acc[i][j] = wmma_bf16(afh[i], bfl[j], acc[i][j]);
                acc[i][j] = wmma_bf16(afl[i], bfh[j], acc[i][j]);
            }
    }

#pragma unroll
    for (int i = 0; i < 2; ++i)
#pragma unroll
        for (int j = 0; j < 4; ++j)
#pragma unroll
            for (int r = 0; r < 8; ++r) {
                int row = bm + wm * 32 + i * 16 + r + half * 8;
                int col = bn + wn * 64 + j * 16 + l15;
                C[(size_t)row * N + col] = acc[i][j][r];
            }
}

// ---------------------------------------------------------------------------
// RoPE in-place on q,k slices of qkv [B*T, 6144]. freqs_cis: [T, 64, 2].
// ---------------------------------------------------------------------------
__global__ __launch_bounds__(256) void rope_kernel(
    float* __restrict__ qkv, const float* __restrict__ freqs)
{
    const int idx = blockIdx.x * blockDim.x + threadIdx.x;
    const int d     = idx & 63;
    const int h     = (idx >> 6) & 15;
    const int which = (idx >> 10) & 1;
    const int bt    = idx >> 11;
    const int t     = bt & 2047;

    const float c = freqs[(t * 64 + d) * 2 + 0];
    const float s = freqs[(t * 64 + d) * 2 + 1];
    float* p = qkv + (size_t)bt * 6144 + which * 2048 + h * 128 + 2 * d;
    float x0 = p[0], x1 = p[1];
    p[0] = x0 * c - x1 * s;
    p[1] = x1 * c + x0 * s;
}

// ---------------------------------------------------------------------------
// Flash attention (causal), exact f32 WMMA path, async double-buffered K/V
// staging (both natural [krow][hd] layout). Writes bf16 hi/lo output.
// Grid: (B*H, T/128). 8 waves; wave w owns q rows [qbase+16w, qbase+16w+16).
// ---------------------------------------------------------------------------
__global__ __launch_bounds__(256) void flash_attn(
    const float* __restrict__ qkv,
    __bf16* __restrict__ attnh, __bf16* __restrict__ attnl)
{
    constexpr int T = 2048, DQKV = 6144;
    __shared__ __align__(16) float KV[2][2][16 * 128]; // [buf][K/V][krow][hd]
    __shared__ __align__(16) float Ps[8 * 16 * 16];    // per-wave P tile [m][k]

    const int tid  = threadIdx.x;
    const int lane = tid & 31;
    const int wave = tid >> 5;
    const int half = lane >> 4;
    const int l15  = lane & 15;
    const int b    = blockIdx.x >> 4;
    const int h    = blockIdx.x & 15;
    const int qbase = blockIdx.y * 128;
    const int qt    = (qbase >> 4) + wave;
    const size_t rowbase = (size_t)b * T;

    // Load this wave's Q tile (16x128) straight into A-fragment registers.
    v2f qf[32];
    {
        const float* qp = qkv + (rowbase + qbase + wave * 16 + l15) * (size_t)DQKV
                              + h * 128 + half * 2;
#pragma unroll
        for (int c = 0; c < 32; ++c) qf[c] = *(const v2f*)(qp + c * 4);
    }

    v8f oacc[8];
#pragma unroll
    for (int j = 0; j < 8; ++j) oacc[j] = (v8f){};
    float m[8], l[8];
#pragma unroll
    for (int r = 0; r < 8; ++r) { m[r] = -3.0e38f; l[r] = 0.0f; }

    // Stage K and V tiles (16x128 f32 each) for k-tile kt into buffer buf.
    auto stage = [&](int kt, int buf) {
#pragma unroll
        for (int i = 0; i < 2; ++i) {
            int e = tid + i * 256;
            int r = e >> 5, c4 = (e & 31) * 4;
            const float* src = qkv + (rowbase + kt * 16 + r) * (size_t)DQKV
                                   + 2048 + h * 128 + c4;
            cp16_g2l(src,        &KV[buf][0][r * 128 + c4]);  // K
            cp16_g2l(src + 2048, &KV[buf][1][r * 128 + c4]);  // V
        }
    };

    const int nkt = (qbase >> 4) + 8;
    stage(0, 0);
    for (int kt = 0; kt < nkt; ++kt) {
        const int buf = kt & 1;
        wait_cp();
        __syncthreads();
        if (kt + 1 < nkt) stage(kt + 1, buf ^ 1);  // overlap DMA with compute

        if (kt <= qt) {
            const float* Ks = KV[buf][0];
            const float* Vs = KV[buf][1];

            // S = Q * K^T  (K-dim = HD = 128 -> 32 WMMAs of K=4)
            v8f S = (v8f){};
#pragma unroll
            for (int c = 0; c < 32; ++c) {
                v2f bf = *(const v2f*)&Ks[l15 * 128 + c * 4 + half * 2];
                S = wmma_f32(qf[c], bf, S);
            }
            const float scale = 0.08838834764831845f;
#pragma unroll
            for (int r = 0; r < 8; ++r) S[r] *= scale;

            if (kt == qt) {                       // causal mask on diagonal tile
                int col = kt * 16 + l15;
#pragma unroll
                for (int r = 0; r < 8; ++r) {
                    int row = qbase + wave * 16 + r + half * 8;
                    if (col > row) S[r] = -3.0e38f;
                }
            }

            // Online softmax per row (16 lanes per half hold one row).
#pragma unroll
            for (int r = 0; r < 8; ++r) {
                float s = S[r];
                float tmax = s;
                tmax = fmaxf(tmax, __shfl_xor(tmax, 8, 16));
                tmax = fmaxf(tmax, __shfl_xor(tmax, 4, 16));
                tmax = fmaxf(tmax, __shfl_xor(tmax, 2, 16));
                tmax = fmaxf(tmax, __shfl_xor(tmax, 1, 16));
                float mn    = fmaxf(m[r], tmax);
                float alpha = __expf(m[r] - mn);
                float p     = __expf(s - mn);
                float rs = p;
                rs += __shfl_xor(rs, 8, 16);
                rs += __shfl_xor(rs, 4, 16);
                rs += __shfl_xor(rs, 2, 16);
                rs += __shfl_xor(rs, 1, 16);
                l[r] = l[r] * alpha + rs;
                m[r] = mn;
#pragma unroll
                for (int j = 0; j < 8; ++j) oacc[j][r] *= alpha;
                Ps[wave * 256 + (r + half * 8) * 16 + l15] = p;   // [m][k]
            }

            // O += P * V. B-fragment from natural V: {V[k][n], V[k+1][n]}
            // -> ds_load_2addr_b32 pair (stride 128 dwords).
#pragma unroll
            for (int c = 0; c < 4; ++c) {
                v2f af = *(const v2f*)&Ps[wave * 256 + l15 * 16 + c * 4 + half * 2];
#pragma unroll
                for (int j = 0; j < 8; ++j) {
                    const float* vp = &Vs[(c * 4 + half * 2) * 128 + j * 16 + l15];
                    v2f bf = { vp[0], vp[128] };
                    oacc[j] = wmma_f32(af, bf, oacc[j]);
                }
            }
        }
    }

    // Normalize; emit bf16 hi/lo so the out-projection GEMM can DMA it.
#pragma unroll
    for (int r = 0; r < 8; ++r) {
        float inv = 1.0f / l[r];
        int row = qbase + wave * 16 + r + half * 8;
        size_t off = (rowbase + row) * (size_t)2048 + h * 128 + l15;
#pragma unroll
        for (int j = 0; j < 8; ++j) {
            float o = oacc[j][r] * inv;
            __bf16 hh = (__bf16)o;
            attnh[off + j * 16] = hh;
            attnl[off + j * 16] = (__bf16)(o - (float)hh);
        }
    }
}

// ---------------------------------------------------------------------------
extern "C" void kernel_launch(void* const* d_in, const int* in_sizes, int n_in,
                              void* d_out, int out_size, void* d_ws, size_t ws_size,
                              hipStream_t stream)
{
    const float* x     = (const float*)d_in[0];  // [4096, 2048]
    const float* Wqkv  = (const float*)d_in[1];  // [2048, 6144]
    const float* Wout  = (const float*)d_in[2];  // [2048, 2048]
    const float* freqs = (const float*)d_in[3];  // [2048, 64, 2]
    float* out = (float*)d_out;                  // [4096, 2048]

    char* ws = (char*)d_ws;
    float*  qkv = (float*)ws;                             // 100.7 MB f32
    __bf16* Xh  = (__bf16*)(ws + (size_t)100663296);      // 16.8 MB
    __bf16* Xl  = (__bf16*)(ws + (size_t)117440512);
    __bf16* ATh = (__bf16*)(ws + (size_t)134217728);      // attn hi/lo
    __bf16* ATl = (__bf16*)(ws + (size_t)150994944);
    __bf16* Wqh = (__bf16*)(ws + (size_t)167772160);      // 25.2 MB
    __bf16* Wql = (__bf16*)(ws + (size_t)192937984);
    __bf16* Woh = (__bf16*)(ws + (size_t)218103808);      //  8.4 MB
    __bf16* Wol = (__bf16*)(ws + (size_t)226492416);      // end 234.9 MB

    // 0) Split activations + weights (weights also transposed to [N,K]).
    split_x  <<<8388608 / 256, 256, 0, stream>>>(x, Xh, Xl);
    split_w_t<<<(6144 * 2048) / 256, 256, 0, stream>>>(Wqkv, Wqh, Wql, 2048, 6144);
    split_w_t<<<(2048 * 2048) / 256, 256, 0, stream>>>(Wout, Woh, Wol, 2048, 2048);

    // 1) qkv = x @ Wqkv  (M=4096, N=6144, K=2048), bf16x3 WMMA + async staging.
    gemm_bf16x3<<<dim3(6144 / 128, 4096 / 128), 256, 0, stream>>>(
        Xh, Xl, Wqh, Wql, qkv, 4096, 6144, 2048);

    // 2) RoPE in place.
    rope_kernel<<<8388608 / 256, 256, 0, stream>>>(qkv, freqs);

    // 3) Causal flash attention (f32 WMMA, async K/V staging) -> bf16 hi/lo.
    flash_attn<<<dim3(32, 16), 256, 0, stream>>>(qkv, ATh, ATl);

    // 4) out = attn @ Wout (M=4096, N=2048, K=2048).
    gemm_bf16x3<<<dim3(2048 / 128, 4096 / 128), 256, 0, stream>>>(
        ATh, ATl, Woh, Wol, out, 4096, 2048, 2048);
}
